// AdaptiveFFNMoE_63513976373307
// MI455X (gfx1250) — matrix operand order repaired and checked by reading, not compiled, not run on hardware
//
#include <hip/hip_runtime.h>
#include <math.h>

#define D     2048
#define E     8
#define TOPK  2
#define HMAX  11264
#define NTOK  8192
#define LN_EPS 1e-5f

typedef __attribute__((ext_vector_type(16))) __bf16 v16bf;
typedef __attribute__((ext_vector_type(8)))  __bf16 bf16x8;
typedef __attribute__((ext_vector_type(8)))  float  v8f;

typedef int v4i_vs __attribute__((vector_size(16)));

#if defined(__gfx1250__) && \
    __has_builtin(__builtin_amdgcn_global_load_async_to_lds_b128) && \
    __has_builtin(__builtin_amdgcn_s_wait_asynccnt)
#define USE_ASYNC_LDS 1
#else
#define USE_ASYNC_LDS 0
#endif

#define GLB128(p) ((__attribute__((address_space(1))) v4i_vs*)(p))
#define LDS128(p) ((__attribute__((address_space(3))) v4i_vs*)(p))

// ---------------------------------------------------------------------------
// Kernel 0: zero the per-expert assignment counters
// ---------------------------------------------------------------------------
__global__ void zero_counts(int* counts) {
    if (threadIdx.x < E) counts[threadIdx.x] = 0;
}

// ---------------------------------------------------------------------------
// Kernel 1: LayerNorm + router softmax + adaptive top-k + assignment build.
// ---------------------------------------------------------------------------
__global__ __launch_bounds__(256)
void moe_prologue(const float* __restrict__ x,
                  const float* __restrict__ ln_g, const float* __restrict__ ln_b,
                  const float* __restrict__ gate_W, const float* __restrict__ gate_b,
                  const float* __restrict__ tpW1, const float* __restrict__ tpb1,
                  const float* __restrict__ tpW2, const float* __restrict__ tpb2,
                  __bf16* __restrict__ xn_out,
                  float* __restrict__ out,
                  int* __restrict__ counts,
                  int* __restrict__ tok_list,
                  float* __restrict__ wgt_list)
{
    __shared__ float xs[D];
    __shared__ float red[256];
    __shared__ float glog[E];
    __shared__ float hid4[4][64];
    __shared__ float hid[64];
    __shared__ float tp2[TOPK];

    const int row = blockIdx.x;
    const int t   = threadIdx.x;
    const float* xr = x + (size_t)row * D;

    float s = 0.f;
    for (int i = t; i < D; i += 256) s += xr[i];
    red[t] = s; __syncthreads();
    for (int o = 128; o > 0; o >>= 1) { if (t < o) red[t] += red[t + o]; __syncthreads(); }
    const float mu = red[0] * (1.0f / D);
    __syncthreads();

    float v = 0.f;
    for (int i = t; i < D; i += 256) { float dl = xr[i] - mu; v += dl * dl; }
    red[t] = v; __syncthreads();
    for (int o = 128; o > 0; o >>= 1) { if (t < o) red[t] += red[t + o]; __syncthreads(); }
    const float rstd = rsqrtf(red[0] * (1.0f / D) + LN_EPS);
    __syncthreads();

    for (int i = t; i < D; i += 256) {
        float xv = xr[i];
        float xn = (xv - mu) * rstd * ln_g[i] + ln_b[i];
        xs[i] = xn;
        xn_out[(size_t)row * D + i] = (__bf16)xn;
        out[(size_t)row * D + i]    = xv;
    }
    __syncthreads();

    const int wave = t >> 5, lane = t & 31;

    if (wave < E) {
        float acc = 0.f;
        for (int d = lane; d < D; d += 32) acc += xs[d] * gate_W[d * E + wave];
        for (int o = 16; o > 0; o >>= 1) acc += __shfl_down(acc, o, 32);
        if (lane == 0) glog[wave] = acc + gate_b[wave];
    }

    {
        const int h = t & 63, c = t >> 6;
        float acc = 0.f;
        const int d0 = c * (D / 4), d1 = d0 + (D / 4);
        for (int d = d0; d < d1; ++d) acc += xs[d] * tpW1[d * 64 + h];
        hid4[c][h] = acc;
    }
    __syncthreads();
    if (t < 64) {
        float a = hid4[0][t] + hid4[1][t] + hid4[2][t] + hid4[3][t] + tpb1[t];
        hid[t] = a > 0.f ? a : 0.f;
    }
    __syncthreads();
    if (t < TOPK) {
        float acc = 0.f;
        for (int i = 0; i < 64; ++i) acc += hid[i] * tpW2[i * TOPK + t];
        tp2[t] = acc + tpb2[t];
    }
    __syncthreads();

    if (t == 0) {
        float tws = 1.f / (1.f + expf(-tp2[0])) + 1.f / (1.f + expf(-tp2[1]));
        int effk = (int)rintf(tws);
        effk = effk < 1 ? 1 : (effk > TOPK ? TOPK : effk);

        float mx = glog[0];
        for (int e = 1; e < E; ++e) mx = fmaxf(mx, glog[e]);
        float p[E], den = 0.f;
        for (int e = 0; e < E; ++e) { p[e] = expf(glog[e] - mx); den += p[e]; }
        const float rden = 1.f / den;
        for (int e = 0; e < E; ++e) p[e] *= rden;

        int i0 = 0;
        for (int e = 1; e < E; ++e) if (p[e] > p[i0]) i0 = e;
        int i1 = (i0 == 0) ? 1 : 0;
        for (int e = 0; e < E; ++e) if (e != i0 && p[e] > p[i1]) i1 = e;

        const float p0 = p[i0];
        const float p1 = (effk >= 2) ? p[i1] : 0.f;
        const float rs = 1.f / (p0 + p1 + 1e-8f);

        int pos0 = atomicAdd(&counts[i0], 1);
        tok_list[i0 * NTOK + pos0] = row;
        wgt_list[i0 * NTOK + pos0] = p0 * rs;
        if (effk >= 2) {
            int pos1 = atomicAdd(&counts[i1], 1);
            tok_list[i1 * NTOK + pos1] = row;
            wgt_list[i1 * NTOK + pos1] = p1 * rs;
        }
    }
}

// ---------------------------------------------------------------------------
// Kernel: fp32 -> bf16 convert + transpose. in: logical [R][C] (row stride
// in_stride), out: [C][R] bf16 (row stride out_stride == R). 32x32 LDS tiles.
// ---------------------------------------------------------------------------
__global__ __launch_bounds__(256)
void cvt_transpose(const float* __restrict__ in, __bf16* __restrict__ out,
                   int in_stride, int out_stride)
{
    __shared__ __bf16 tile[32][33];
    const int c0 = blockIdx.x * 32, r0 = blockIdx.y * 32;
    const int t = threadIdx.x;
#pragma unroll
    for (int it = 0; it < 4; ++it) {
        int idx = t + it * 256;
        int rr = idx >> 5, cc = idx & 31;
        tile[rr][cc] = (__bf16)in[(size_t)(r0 + rr) * in_stride + c0 + cc];
    }
    __syncthreads();
#pragma unroll
    for (int it = 0; it < 4; ++it) {
        int idx = t + it * 256;
        int cc = idx >> 5, rr = idx & 31;
        out[(size_t)(c0 + cc) * out_stride + r0 + rr] = tile[rr][cc];
    }
}

// ---------------------------------------------------------------------------
// WMMA fragment loads (ISA §7.12.2, wave32). Tiles row-major along K,
// row stride 72 halves (144 B). kk = 0 or 32.
// ---------------------------------------------------------------------------
__device__ inline v16bf load_frag_a(const __bf16* As, int lane, int kk) {
    const int m  = lane & 15;
    const int kb = kk + ((lane < 16) ? 0 : 8);
    bf16x8 a0 = *(const bf16x8*)(As + m * 72 + kb);
    bf16x8 a1 = *(const bf16x8*)(As + m * 72 + kb + 16);
    v16bf a;
#pragma unroll
    for (int i = 0; i < 8; ++i) { a[i] = a0[i]; a[i + 8] = a1[i]; }
    return a;
}

__device__ inline v16bf load_frag_b(const __bf16* Bt, int lane, int kk) {
    const int n  = lane & 15;
    const int kb = kk + ((lane < 16) ? 0 : 8);
    bf16x8 b0 = *(const bf16x8*)(Bt + n * 72 + kb);
    bf16x8 b1 = *(const bf16x8*)(Bt + n * 72 + kb + 16);
    v16bf b;
#pragma unroll
    for (int i = 0; i < 8; ++i) { b[i] = b0[i]; b[i + 8] = b1[i]; }
    return b;
}

// Issue one K-chunk stage (A: 1 x b128 per thread; B: 4 x b128 per lane).
// 5 async instructions per wave per stage.
__device__ inline void stage_issue(const __bf16* ag, __bf16* al,
                                   const __bf16* bg, __bf16* bl)
{
#if USE_ASYNC_LDS
    __builtin_amdgcn_global_load_async_to_lds_b128(GLB128(ag), LDS128(al), 0, 0);
#pragma unroll
    for (int j = 0; j < 4; ++j)
        __builtin_amdgcn_global_load_async_to_lds_b128(GLB128(bg + j * 8),
                                                       LDS128(bl + j * 8), 0, 0);
#else
    *(uint4*)al = *(const uint4*)ag;
#pragma unroll
    for (int j = 0; j < 4; ++j)
        *(uint4*)(bl + j * 8) = *(const uint4*)(bg + j * 8);
#endif
}

// 4 WMMAs per stage: two M-subtiles (rows 0-15, 16-31) sharing one B frag.
__device__ inline void wmma_stage(const __bf16 (*As)[72], const __bf16 (*Bsw)[72],
                                  int lane, v8f* cacc)
{
#pragma unroll
    for (int kk = 0; kk < 64; kk += 32) {
        v16bf b  = load_frag_b(&Bsw[0][0], lane, kk);
        v16bf a0 = load_frag_a(&As[0][0],  lane, kk);
        v16bf a1 = load_frag_a(&As[16][0], lane, kk);
        cacc[0] = __builtin_amdgcn_wmma_f32_16x16x32_bf16(false, a0, false, b,
                                                          (short)0, cacc[0], false, false);
        cacc[1] = __builtin_amdgcn_wmma_f32_16x16x32_bf16(false, a1, false, b,
                                                          (short)0, cacc[1], false, false);
    }
}

// ---------------------------------------------------------------------------
// Kernel: up-projection  h = GELU(xn[toks] @ Wi_t^T + bi).  Wi_t: [H][D] bf16.
// Block = 8 waves: 32 tokens x 128 h-cols, K-chunks of 64, double-buffered
// async prefetch.
// ---------------------------------------------------------------------------
__global__ __launch_bounds__(256)
void moe_up(const __bf16* __restrict__ xn,
            const __bf16* __restrict__ Wi_t,
            const float* __restrict__ bi,
            const int* __restrict__ counts, const int* __restrict__ tok_list,
            __bf16* __restrict__ hbuf,
            int expert, int H)
{
    const int cnt = counts[expert];
    const int mtile = blockIdx.x;
    if (mtile * 32 >= cnt) return;

    const int hb   = blockIdx.y * 128;
    const int t    = threadIdx.x;
    const int wave = t >> 5, lane = t & 31;

    __shared__ __bf16 As[2][32][72];
    __shared__ __bf16 Bs[2][8][16][72];

    // Hoisted per-thread source pointers (constant across the K loop)
    const int ar = t >> 3, ac = (t & 7) * 8;            // A: row 0..31, 8 halves
    int pa = mtile * 32 + ar; if (pa >= cnt) pa = cnt - 1;
    const int tok_a = tok_list[expert * NTOK + pa];
    const __bf16* ag = xn + (size_t)tok_a * D + ac;

    const int br = lane & 15, bh = (lane >> 4) * 32;    // B: row 0..15, 32 halves
    const __bf16* bg = Wi_t + (size_t)(hb + wave * 16 + br) * D + bh;

    v8f cacc[2] = {};

    stage_issue(ag, &As[0][ar][ac], bg, &Bs[0][wave][br][bh]);
    const int nk = D / 64;
    for (int kc = 0; kc < nk; ++kc) {
        const int cur = kc & 1;
        if (kc + 1 < nk) {
            const int k0n = (kc + 1) * 64;
            stage_issue(ag + k0n, &As[1 - cur][ar][ac],
                        bg + k0n, &Bs[1 - cur][wave][br][bh]);
#if USE_ASYNC_LDS
            __builtin_amdgcn_s_wait_asynccnt(5);   // current stage resident
#endif
        }
#if USE_ASYNC_LDS
        else { __builtin_amdgcn_s_wait_asynccnt(0); }
#endif
        __syncthreads();
        wmma_stage(As[cur], Bs[cur][wave], lane, cacc);
        __syncthreads();
    }

    const int n  = lane & 15;
    const int mo = (lane < 16) ? 0 : 8;
    const int hcol = hb + wave * 16 + n;
    const float bv = bi[expert * HMAX + hcol];
#pragma unroll
    for (int half = 0; half < 2; ++half) {
#pragma unroll
        for (int r = 0; r < 8; ++r) {
            int p = mtile * 32 + half * 16 + r + mo;
            if (p < cnt) {
                float hv = cacc[half][r] + bv;
                float g  = 0.5f * hv * (1.f + erff(hv * 0.70710678118654752f));
                hbuf[(size_t)p * H + hcol] = (__bf16)g;
            }
        }
    }
}

// ---------------------------------------------------------------------------
// Kernel: down-projection + weighted scatter-accumulate.
// Wo_t: [D][H] bf16.  out[tok] += w * (h @ Wo_t^T + bo).
// ---------------------------------------------------------------------------
__global__ __launch_bounds__(256)
void moe_down(const __bf16* __restrict__ hbuf,
              const __bf16* __restrict__ Wo_t,
              const float* __restrict__ bo,
              const int* __restrict__ counts, const int* __restrict__ tok_list,
              const float* __restrict__ wgt_list,
              float* __restrict__ out,
              int expert, int H)
{
    const int cnt = counts[expert];
    const int mtile = blockIdx.x;
    if (mtile * 32 >= cnt) return;

    const int db   = blockIdx.y * 128;
    const int t    = threadIdx.x;
    const int wave = t >> 5, lane = t & 31;

    __shared__ __bf16 As[2][32][72];
    __shared__ __bf16 Bs[2][8][16][72];

    const int ar = t >> 3, ac = (t & 7) * 8;
    int pa = mtile * 32 + ar; if (pa >= cnt) pa = cnt - 1;
    const __bf16* ag = hbuf + (size_t)pa * H + ac;

    const int br = lane & 15, bh = (lane >> 4) * 32;
    const __bf16* bg = Wo_t + (size_t)(db + wave * 16 + br) * H + bh;

    v8f cacc[2] = {};

    stage_issue(ag, &As[0][ar][ac], bg, &Bs[0][wave][br][bh]);
    const int nk = H / 64;
    for (int kc = 0; kc < nk; ++kc) {
        const int cur = kc & 1;
        if (kc + 1 < nk) {
            const int k0n = (kc + 1) * 64;
            stage_issue(ag + k0n, &As[1 - cur][ar][ac],
                        bg + k0n, &Bs[1 - cur][wave][br][bh]);
#if USE_ASYNC_LDS
            __builtin_amdgcn_s_wait_asynccnt(5);
#endif
        }
#if USE_ASYNC_LDS
        else { __builtin_amdgcn_s_wait_asynccnt(0); }
#endif
        __syncthreads();
        wmma_stage(As[cur], Bs[cur][wave], lane, cacc);
        __syncthreads();
    }

    const int n  = lane & 15;
    const int mo = (lane < 16) ? 0 : 8;
    const int dcol = db + wave * 16 + n;
    const float bv = bo[expert * D + dcol];
#pragma unroll
    for (int half = 0; half < 2; ++half) {
#pragma unroll
        for (int r = 0; r < 8; ++r) {
            int p = mtile * 32 + half * 16 + r + mo;
            if (p < cnt) {
                int   tok = tok_list[expert * NTOK + p];
                float w   = wgt_list[expert * NTOK + p];
                float* o  = out + (size_t)tok * D + dcol;
                *o += w * (cacc[half][r] + bv);
            }
        }
    }
}

// ---------------------------------------------------------------------------
// Host launcher
// ---------------------------------------------------------------------------
extern "C" void kernel_launch(void* const* d_in, const int* in_sizes, int n_in,
                              void* d_out, int out_size, void* d_ws, size_t ws_size,
                              hipStream_t stream) {
    const float* x      = (const float*)d_in[0];
    const float* ln_g   = (const float*)d_in[1];
    const float* ln_b   = (const float*)d_in[2];
    const float* gate_W = (const float*)d_in[3];
    const float* gate_b = (const float*)d_in[4];
    const float* tpW1   = (const float*)d_in[5];
    const float* tpb1   = (const float*)d_in[6];
    const float* tpW2   = (const float*)d_in[7];
    const float* tpb2   = (const float*)d_in[8];
    const float* Wi     = (const float*)d_in[9];
    const float* bi     = (const float*)d_in[10];
    const float* Wo     = (const float*)d_in[11];
    const float* bo     = (const float*)d_in[12];
    float* out = (float*)d_out;

    char* ws = (char*)d_ws;
    size_t off = 0;
    int*    counts   = (int*)(ws + off);    off += 256;
    int*    tok_list = (int*)(ws + off);    off += (size_t)E * NTOK * 4;
    float*  wgt_list = (float*)(ws + off);  off += (size_t)E * NTOK * 4;
    __bf16* xn       = (__bf16*)(ws + off); off += (size_t)NTOK * D * 2;
    __bf16* hbuf     = (__bf16*)(ws + off); off += (size_t)NTOK * HMAX * 2;
    __bf16* Wi_t     = (__bf16*)(ws + off); off += (size_t)HMAX * D * 2;  // [H][D]
    __bf16* Wo_t     = (__bf16*)(ws + off); off += (size_t)D * HMAX * 2;  // [D][H]

    zero_counts<<<1, 64, 0, stream>>>(counts);
    moe_prologue<<<NTOK, 256, 0, stream>>>(x, ln_g, ln_b, gate_W, gate_b,
                                           tpW1, tpb1, tpW2, tpb2,
                                           xn, out, counts, tok_list, wgt_list);

    static const int HS[E] = {4096, 5120, 6144, 7168, 8192, 9216, 10240, 11264};
    for (int e = 0; e < E; ++e) {
        const int H = HS[e];
        cvt_transpose<<<dim3(H / 32, D / 32), 256, 0, stream>>>(
            Wi + (size_t)e * D * HMAX, Wi_t, HMAX, D);
        cvt_transpose<<<dim3(D / 32, H / 32), 256, 0, stream>>>(
            Wo + (size_t)e * HMAX * D, Wo_t, D, H);

        dim3 gu(NTOK / 32, H / 128);
        moe_up<<<gu, 256, 0, stream>>>(xn, Wi_t, bi, counts, tok_list, hbuf, e, H);
        dim3 gd(NTOK / 32, D / 128);
        moe_down<<<gd, 256, 0, stream>>>(hbuf, Wo_t, bo, counts, tok_list, wgt_list,
                                         out, e, H);
    }
}
